// FourierMeshGraphNet_38345468018713
// MI455X (gfx1250) — matrix operand order repaired and checked by reading, not compiled, not run on hardware
//
#include <hip/hip_runtime.h>
#include <hip/hip_bf16.h>
#include <math.h>

typedef __attribute__((ext_vector_type(16))) _Float16 v16h;
typedef __attribute__((ext_vector_type(8)))  float    v8f;

#define NNODES 10000
#define NEDGES 100000
#define HDIM   128
#define BLK_M  32
#define A_STRIDE 392   // halfs; 784B row stride: 16B aligned, lane-distinct banks
#define H_STRIDE 136   // halfs; 272B row stride: 16B aligned
#define C_STRIDE 129   // f32; stride%64==1 -> conflict-free LN row scans

// ---- A-tile segment stager: vectorized when C%4==0, scalar fallback ----
template<int C>
__device__ __forceinline__ void stage_seg(_Float16* Atile, const float* __restrict__ src,
                                          const int* __restrict__ idx, int colOff,
                                          int m0, int M, int tid) {
    if constexpr ((C & 3) == 0) {
        constexpr int V = C / 4;
        #pragma unroll
        for (int e = tid; e < BLK_M * V; e += 256) {
            int r = e / V, c4 = (e - r * V) * 4;
            int gm = m0 + r;
            float4 v = make_float4(0.f, 0.f, 0.f, 0.f);
            if (gm < M) {
                int rr = idx ? idx[gm] : gm;
                v = *(const float4*)(src + (size_t)rr * C + c4);
            }
            union { _Float16 h[4]; short4 s4; } u;
            u.h[0] = (_Float16)v.x; u.h[1] = (_Float16)v.y;
            u.h[2] = (_Float16)v.z; u.h[3] = (_Float16)v.w;
            *(short4*)&Atile[r * A_STRIDE + colOff + c4] = u.s4;
        }
    } else {
        for (int e = tid; e < BLK_M * C; e += 256) {
            int r = e / C, c = e - r * C;
            int gm = m0 + r;
            float v = 0.f;
            if (gm < M) {
                int rr = idx ? idx[gm] : gm;
                v = src[(size_t)rr * C + c];
            }
            Atile[r * A_STRIDE + colOff + c] = (_Float16)v;
        }
    }
}

// ---------------------------------------------------------------------------
// Fused 2-layer MLP: out = [LN](relu(A@W0+b0) @ W1 + b1) [+ resid]
// A = concat(src0, src1[idx1], src2[idx2]); optional fused atomic segment-sum.
// Weights are pre-transposed f16: W0t[128][KPAD], W1t[128][128] (zero padded).
// B fragments are read directly from global (L2-resident) -> no K-loop barriers.
// ---------------------------------------------------------------------------
template<int C0, int C1, int C2, int N2, bool LN>
__global__ __launch_bounds__(256) void mlp2_kernel(
    const float* __restrict__ src0,
    const float* __restrict__ src1, const int* __restrict__ idx1,
    const float* __restrict__ src2, const int* __restrict__ idx2,
    int M,
    const _Float16* __restrict__ W0t, const float* __restrict__ b0,  // [128][KPAD]
    const _Float16* __restrict__ W1t, const float* __restrict__ b1,  // [128][128]
    const float* __restrict__ ln_g, const float* __restrict__ ln_b,
    const float* __restrict__ resid,
    float* __restrict__ out,
    float* __restrict__ agg, const int* __restrict__ aggIdx)
{
    constexpr int K    = C0 + C1 + C2;
    constexpr int NCH  = (K + 31) / 32;
    constexpr int KPAD = NCH * 32;

    __shared__ _Float16 Atile[BLK_M * A_STRIDE];
    __shared__ _Float16 Htile[BLK_M * H_STRIDE];
    __shared__ float    Cbuf [BLK_M * C_STRIDE];
    __shared__ float    rowMu[BLK_M], rowRs[BLK_M];

    const int tid  = threadIdx.x;
    const int lane = tid & 31;
    const int wv   = tid >> 5;
    const int rblk = wv & 1;    // 16-row block within 32 rows
    const int cblk = wv >> 1;   // 32-col block within 128 cols
    const int m0   = blockIdx.x * BLK_M;

    // ---- stage A tile (fused gather + f32->f16) ----
    stage_seg<C0>(Atile, src0, nullptr, 0, m0, M, tid);
    if constexpr (C1 > 0) stage_seg<C1>(Atile, src1, idx1, C0, m0, M, tid);
    if constexpr (C2 > 0) stage_seg<C2>(Atile, src2, idx2, C0 + C1, m0, M, tid);
    if constexpr (KPAD > K) {
        constexpr int PADC = KPAD - K;
        for (int e = tid; e < BLK_M * PADC; e += 256) {
            int r = e / PADC, c = e - r * PADC;
            Atile[r * A_STRIDE + K + c] = (_Float16)0.f;
        }
    }
    __syncthreads();

    const v8f vzero = {};
    v8f acc0 = vzero, acc1 = vzero;

    const int aRow = (rblk * 16 + (lane & 15)) * A_STRIDE + ((lane >> 4) << 3);
    const int n0 = cblk * 32 + (lane & 15);              // fragment column (t adds 16)
    const int kg = (lane >> 4) << 3;                     // B fragment K offset

    // ---- GEMM1: hidden = relu(A @ W0 + b0), N = 128; no barriers in K-loop ----
    #pragma unroll
    for (int kc = 0; kc < NCH; ++kc) {
        union { v16h v; uint4 q[2]; } af;                // ISA 16-bit A 16x32 layout
        const _Float16* ap = &Atile[aRow + (kc << 5)];
        af.q[0] = *(const uint4*)ap;
        af.q[1] = *(const uint4*)(ap + 16);
        #pragma unroll
        for (int t = 0; t < 2; ++t) {
            union { v16h v; uint4 q[2]; } bf;            // ISA 16-bit B 32x16 layout
            const _Float16* bp = W0t + (size_t)(n0 + t * 16) * KPAD + (kc << 5) + kg;
            bf.q[0] = *(const uint4*)bp;
            bf.q[1] = *(const uint4*)(bp + 16);
            v8f& acc = t ? acc1 : acc0;
            acc = __builtin_amdgcn_wmma_f32_16x16x32_f16(
                      false, af.v, false, bf.v, (short)0, acc, false, false);
        }
    }

    // bias + relu -> Htile (f16)
    #pragma unroll
    for (int t = 0; t < 2; ++t) {
        v8f acc = t ? acc1 : acc0;
        int n = n0 + t * 16;
        #pragma unroll
        for (int v = 0; v < 8; ++v) {
            int m = rblk * 16 + v + 8 * (lane >> 4);     // ISA C/D layout
            float x = acc[v] + b0[n];
            Htile[m * H_STRIDE + n] = (_Float16)(x > 0.f ? x : 0.f);
        }
    }
    __syncthreads();

    // ---- GEMM2: C = hidden @ W1 + b1 (K = 128) ----
    acc0 = vzero; acc1 = vzero;
    const int hRow = (rblk * 16 + (lane & 15)) * H_STRIDE + ((lane >> 4) << 3);
    #pragma unroll
    for (int kc = 0; kc < 4; ++kc) {
        union { v16h v; uint4 q[2]; } af;
        const _Float16* ap = &Htile[hRow + (kc << 5)];
        af.q[0] = *(const uint4*)ap;
        af.q[1] = *(const uint4*)(ap + 16);
        #pragma unroll
        for (int t = 0; t < 2; ++t) {
            union { v16h v; uint4 q[2]; } bf;
            const _Float16* bp = W1t + (size_t)(n0 + t * 16) * HDIM + (kc << 5) + kg;
            bf.q[0] = *(const uint4*)bp;
            bf.q[1] = *(const uint4*)(bp + 16);
            v8f& acc = t ? acc1 : acc0;
            acc = __builtin_amdgcn_wmma_f32_16x16x32_f16(
                      false, af.v, false, bf.v, (short)0, acc, false, false);
        }
    }

    #pragma unroll
    for (int t = 0; t < 2; ++t) {
        v8f acc = t ? acc1 : acc0;
        int n = n0 + t * 16;
        #pragma unroll
        for (int v = 0; v < 8; ++v) {
            int m = rblk * 16 + v + 8 * (lane >> 4);
            Cbuf[m * C_STRIDE + n] = acc[v] + ((n < N2) ? b1[n] : 0.f);
        }
    }
    __syncthreads();

    // ---- LayerNorm statistics over 128 features ----
    if constexpr (LN) {
        if (tid < BLK_M) {
            float s = 0.f, ss = 0.f;
            for (int n = 0; n < HDIM; ++n) {
                float x = Cbuf[tid * C_STRIDE + n];
                s += x; ss += x * x;
            }
            float mu  = s * (1.f / HDIM);
            float var = ss * (1.f / HDIM) - mu * mu;
            rowMu[tid] = mu;
            rowRs[tid] = rsqrtf(var + 1e-5f);
        }
        __syncthreads();
    }

    // ---- epilogue: LN, residual, store, fused atomic segment-sum ----
    #pragma unroll
    for (int e = tid; e < BLK_M * N2; e += 256) {
        int r = e / N2, n = e - r * N2;
        int gm = m0 + r;
        if (gm >= M) continue;
        float x = Cbuf[r * C_STRIDE + n];
        if constexpr (LN) x = (x - rowMu[r]) * rowRs[r] * ln_g[n] + ln_b[n];
        if (resid) x += resid[(size_t)gm * N2 + n];
        out[(size_t)gm * N2 + n] = x;
        if (agg) atomicAdd(&agg[(size_t)aggIdx[gm] * HDIM + n], x);
    }
}

// Transpose + f32->f16 weight prep: W[K][N] -> Wt[NP][KP], zero padded.
__global__ void xpose_kernel(const float* __restrict__ W, _Float16* __restrict__ Wt,
                             int K, int N, int KP, int NP, int total) {
    int i = blockIdx.x * blockDim.x + threadIdx.x;
    if (i >= total) return;
    int n = i / KP, k = i - n * KP;
    float v = (k < K && n < N) ? W[(size_t)k * N + n] : 0.f;
    Wt[i] = (_Float16)v;
}

// Fourier features: X[:,0:7]=node_attr; cols 7+d*14+f = cos(x_d*2^(f-3)*pi), +7 = sin
__global__ void fourier_kernel(const float* __restrict__ na, float* __restrict__ Xn) {
    int i = blockIdx.x * blockDim.x + threadIdx.x;
    if (i >= NNODES) return;
    const float* a = na + i * 7;
    float* o = Xn + i * 49;
    #pragma unroll
    for (int k = 0; k < 7; ++k) o[k] = a[k];
    const float PI = 3.14159265358979323846f;
    #pragma unroll
    for (int d = 0; d < 3; ++d) {
        float x = a[d];
        #pragma unroll
        for (int f = 0; f < 7; ++f) {
            float z = x * exp2f((float)(f - 3)) * PI;
            o[7 + d * 14 + f]     = cosf(z);
            o[7 + d * 14 + 7 + f] = sinf(z);
        }
    }
}

__global__ void zero_kernel(float* __restrict__ p, int n) {
    int i = blockIdx.x * blockDim.x + threadIdx.x;
    if (i < n) p[i] = 0.f;
}

extern "C" void kernel_launch(void* const* d_in, const int* in_sizes, int n_in,
                              void* d_out, int out_size, void* d_ws, size_t ws_size,
                              hipStream_t stream) {
    (void)in_sizes; (void)n_in; (void)out_size; (void)ws_size;

    const float* node_attr = (const float*)d_in[0];
    const float* edge_attr = (const float*)d_in[1];
    const int*   ei        = (const int*)d_in[2];
    const int*   srcIdx    = ei;            // edge_index[0]
    const int*   dstIdx    = ei + NEDGES;   // edge_index[1]

    // Pytree leaf order: params{dec, edge_enc, layers[15]{edge,node}, node_enc},
    // each MLP: W0, W1, b0, b1, [ln_b, ln_g]
    auto P = [&](int i) { return (const float*)d_in[i]; };
    const int DEC = 3, EE = 7, LBASE = 13, NE = 193;

    size_t off = 0;
    auto alloc = [&](size_t bytes) {
        void* p = (char*)d_ws + off;
        off = (off + bytes + 255) & ~(size_t)255;
        return p;
    };
    float* Xn  = (float*)alloc((size_t)NNODES * 49 * 4);
    float* nfe = (float*)alloc((size_t)NNODES * HDIM * 4);
    float* efe = (float*)alloc((size_t)NEDGES * HDIM * 4);
    float* agg = (float*)alloc((size_t)NNODES * HDIM * 4);

    // weight prep: transpose + convert, zero-padded to [128][KP]
    auto xpose = [&](const float* W, int K, int N, int KP) -> const _Float16* {
        _Float16* dst = (_Float16*)alloc((size_t)HDIM * KP * 2);
        int total = HDIM * KP;
        xpose_kernel<<<(total + 255) / 256, 256, 0, stream>>>(W, dst, K, N, KP, HDIM, total);
        return dst;
    };

    const _Float16* neW0 = xpose(P(NE), 49, HDIM, 64);
    const _Float16* neW1 = xpose(P(NE + 1), HDIM, HDIM, HDIM);
    const _Float16* eeW0 = xpose(P(EE), 4, HDIM, 32);
    const _Float16* eeW1 = xpose(P(EE + 1), HDIM, HDIM, HDIM);
    const _Float16* dW0  = xpose(P(DEC), HDIM, HDIM, HDIM);
    const _Float16* dW1  = xpose(P(DEC + 1), HDIM, 4, HDIM);
    const _Float16 *leW0[15], *leW1[15], *lnW0[15], *lnW1[15];
    for (int i = 0; i < 15; ++i) {
        const int L = LBASE + 12 * i;
        leW0[i] = xpose(P(L),     3 * HDIM, HDIM, 3 * HDIM);
        leW1[i] = xpose(P(L + 1), HDIM, HDIM, HDIM);
        lnW0[i] = xpose(P(L + 6), 2 * HDIM, HDIM, 2 * HDIM);
        lnW1[i] = xpose(P(L + 7), HDIM, HDIM, HDIM);
    }

    const int NB_N = (NNODES + BLK_M - 1) / BLK_M;   // 313
    const int NB_E = (NEDGES + BLK_M - 1) / BLK_M;   // 3125

    fourier_kernel<<<(NNODES + 255) / 256, 256, 0, stream>>>(node_attr, Xn);

    // node encoder: [49 -> 128 -> 128] + LN
    mlp2_kernel<49, 0, 0, HDIM, true><<<NB_N, 256, 0, stream>>>(
        Xn, nullptr, nullptr, nullptr, nullptr, NNODES,
        neW0, P(NE + 2), neW1, P(NE + 3),
        P(NE + 5), P(NE + 4), nullptr, nfe, nullptr, nullptr);

    // edge encoder: [4 -> 128 -> 128] + LN
    mlp2_kernel<4, 0, 0, HDIM, true><<<NB_E, 256, 0, stream>>>(
        edge_attr, nullptr, nullptr, nullptr, nullptr, NEDGES,
        eeW0, P(EE + 2), eeW1, P(EE + 3),
        P(EE + 5), P(EE + 4), nullptr, efe, nullptr, nullptr);

    for (int i = 0; i < 15; ++i) {
        const int L  = LBASE + 12 * i;   // edge MLP leaves
        const int Ln = L + 6;            // node MLP leaves

        zero_kernel<<<(NNODES * HDIM + 255) / 256, 256, 0, stream>>>(agg, NNODES * HDIM);

        // edge update: e += LN(MLP([e, n[src], n[dst]])), fused scatter-add to agg[dst]
        mlp2_kernel<HDIM, HDIM, HDIM, HDIM, true><<<NB_E, 256, 0, stream>>>(
            efe, nfe, srcIdx, nfe, dstIdx, NEDGES,
            leW0[i], P(L + 2), leW1[i], P(L + 3),
            P(L + 5), P(L + 4), efe, efe, agg, dstIdx);

        // node update: n += LN(MLP([n, agg]))
        mlp2_kernel<HDIM, HDIM, 0, HDIM, true><<<NB_N, 256, 0, stream>>>(
            nfe, agg, nullptr, nullptr, nullptr, NNODES,
            lnW0[i], P(Ln + 2), lnW1[i], P(Ln + 3),
            P(Ln + 5), P(Ln + 4), nfe, nfe, nullptr, nullptr);
    }

    // decoder: [128 -> 128 -> 4], no LN, no residual
    mlp2_kernel<HDIM, 0, 0, 4, false><<<NB_N, 256, 0, stream>>>(
        nfe, nullptr, nullptr, nullptr, nullptr, NNODES,
        dW0, P(DEC + 2), dW1, P(DEC + 3),
        nullptr, nullptr, nullptr, (float*)d_out, nullptr, nullptr);
}
